// UpDownTreeEncoder_33758442946892
// MI455X (gfx1250) — matrix operand order repaired and checked by reading, not compiled, not run on hardware
//
#include <hip/hip_runtime.h>
#include <hip/hip_bf16.h>

typedef __attribute__((ext_vector_type(16))) _Float16 v16h;
typedef __attribute__((ext_vector_type(8)))  float    v8f;

#define DIM    128
#define NRES   262144
#define NTREES 4
#define MTILES_PER_WAVE 4                 // T: m-tiles (of 16 parents) per wave

// workspace layout (bytes)
#define WF_OFF 0u                         // 320KB of f16 B-fragments (padded)
#define HA_OFF 0x80000u                   // h ping buffer: 64MB
#define CA_OFF (HA_OFF + 0x4000000u)      // c ping buffer: 64MB
#define HB_OFF (HA_OFF + 0x8000000u)      // h pong buffer: 32MB
#define CB_OFF (HB_OFF + 0x2000000u)      // c pong buffer: 32MB

__device__ __forceinline__ float rcp_fast(float x) { return __builtin_amdgcn_rcpf(x); }
__device__ __forceinline__ float sigmoidf_fast(float x) {
    return rcp_fast(1.0f + __expf(-x));               // v_exp + v_rcp
}
__device__ __forceinline__ float tanhf_fast(float x) {
    return 1.0f - 2.0f * rcp_fast(1.0f + __expf(2.0f * x));
}

// Repack W_up [640,256] f32 into WMMA B-fragments (f16).
// B[k][n] = W[n][k].  Tile (nt,kt): 16 N cols x 32 K rows.
// Lane L slot j holds B[kt*32 + 16*(L/16) + j][nt*16 + L%16].
__global__ void pack_w_kernel(const float* __restrict__ W, _Float16* __restrict__ wfrag) {
    int tile = blockIdx.x;                       // 0..319 = nt*8 + kt
    int lane = threadIdx.x;                      // 0..31
    int n  = ((tile >> 3) << 4) + (lane & 15);
    int k0 = ((tile & 7) << 5) + ((lane >> 4) << 4);
    const float* src = W + (size_t)n * 256 + k0;
    _Float16* dst = wfrag + ((size_t)tile * 32 + lane) * 16;
#pragma unroll
    for (int j = 0; j < 16; ++j) dst[j] = (_Float16)src[j];
}

// One tree-LSTM up level.  h_in/c_in: [2*Mtot, 128] (children), outputs: [Mtot, 128].
// combined row for parent P is the contiguous 256 floats h_in + P*256.
// Grid: (ceil(Mtot/512), 8).  blockIdx.y = d-tile shared by the whole block.
// Block = 8 waves; B fragments for this d-tile staged once in 40KB LDS; each
// wave then runs MTILES_PER_WAVE m-tiles of 16 parents against LDS-resident B.
__global__ __launch_bounds__(256)
void tree_lstm_level_kernel(const float* __restrict__ h_in,
                            const float* __restrict__ c_in,   // nullptr at leaf level (c==0)
                            const _Float16* __restrict__ wfrag,
                            const float* __restrict__ bias,   // [640]
                            float* __restrict__ h_out,
                            float* __restrict__ c_out,
                            int Mtot)
{
    __shared__ _Float16 bsh[5 * 4096];           // 5 gates x 8 kt x 32 lanes x 16 halfs = 40KB

    const int lane = threadIdx.x & 31;
    const int wid  = threadIdx.x >> 5;           // wave 0..7
    const int dt   = blockIdx.y;                 // d-tile 0..7 (shared by block)
    const int hi   = lane >> 4;                  // lane half
    const int mrow = lane & 15;
    const int d    = dt * 16 + mrow;             // output dim 0..127 (N within gate)

    // ---- cooperative LDS fill: 5 contiguous 8KB chunks (one per gate) ----
#pragma unroll
    for (int g = 0; g < 5; ++g) {
        const uint4* s   = (const uint4*)(wfrag + (size_t)(g * 8 + dt) * 4096);
        uint4*       dst = (uint4*)(bsh + g * 4096);
#pragma unroll
        for (int i = 0; i < 2; ++i)
            dst[threadIdx.x + 256 * i] = s[threadIdx.x + 256 * i];
    }
    __syncthreads();

    const _Float16* bl = bsh + (size_t)lane * 16;   // lane's slot within any fragment

    // bias values for this lane's output dim (uniform across m-tiles)
    const float bi  = bias[d];
    const float bfl = bias[128 + d];
    const float bfr = bias[256 + d];
    const float bo  = bias[384 + d];
    const float bg  = bias[512 + d];

    for (int t = 0; t < MTILES_PER_WAVE; ++t) {
        const int m_base = (blockIdx.x * 8 + wid) * (MTILES_PER_WAVE * 16) + t * 16;
        if (m_base >= Mtot) break;

        // ---- load A fragments: 16 rows x 256 K, f32 -> f16, contiguous k = 16*(lane/16) + s (matches packed B) ----
        v16h a[8];
        {
            const int  P     = m_base + mrow;
            const bool valid = (P < Mtot);
            const float* arow = h_in + (size_t)P * 256;
#pragma unroll
            for (int kt = 0; kt < 8; ++kt) {
                float4 u0 = make_float4(0.f, 0.f, 0.f, 0.f);
                float4 u1 = u0, u2 = u0, u3 = u0;
                if (valid) {
                    const float* p = arow + kt * 32 + hi * 16;
                    u0 = *(const float4*)(p);
                    u1 = *(const float4*)(p + 4);
                    u2 = *(const float4*)(p + 8);
                    u3 = *(const float4*)(p + 12);
                }
                v16h tv;
                tv[0]  = (_Float16)u0.x; tv[1]  = (_Float16)u0.y; tv[2]  = (_Float16)u0.z; tv[3]  = (_Float16)u0.w;
                tv[4]  = (_Float16)u1.x; tv[5]  = (_Float16)u1.y; tv[6]  = (_Float16)u1.z; tv[7]  = (_Float16)u1.w;
                tv[8]  = (_Float16)u2.x; tv[9]  = (_Float16)u2.y; tv[10] = (_Float16)u2.z; tv[11] = (_Float16)u2.w;
                tv[12] = (_Float16)u3.x; tv[13] = (_Float16)u3.y; tv[14] = (_Float16)u3.z; tv[15] = (_Float16)u3.w;
                a[kt] = tv;
            }
        }

        // ---- 8 K-steps x 5 gates of WMMA against LDS-resident B ----
        v8f zero = {};
        v8f acc[5] = {zero, zero, zero, zero, zero};
#pragma unroll
        for (int kt = 0; kt < 8; ++kt) {
#pragma unroll
            for (int g = 0; g < 5; ++g) {
                v16h bf = *(const v16h*)(bl + g * 4096 + kt * 512);
                acc[g] = __builtin_amdgcn_wmma_f32_16x16x32_f16(
                    false, a[kt], false, bf, (short)0, acc[g], false, false);
            }
        }

        // ---- fused LSTM epilogue ----
#pragma unroll
        for (int r = 0; r < 8; ++r) {
            const int Pout = m_base + r + 8 * hi;   // C/D layout: VGPR r -> M = r + 8*(lane/16)
            if (Pout < Mtot) {
                float vi  = sigmoidf_fast(acc[0][r] + bi);
                float vfl = sigmoidf_fast(acc[1][r] + bfl);
                float vfr = sigmoidf_fast(acc[2][r] + bfr);
                float vo  = sigmoidf_fast(acc[3][r] + bo);
                float vg  = tanhf_fast(acc[4][r] + bg);
                float cl = 0.f, cr = 0.f;
                if (c_in) {
                    cl = c_in[(size_t)Pout * 256 + d];
                    cr = c_in[(size_t)Pout * 256 + 128 + d];
                }
                float cn = vfl * cl + vfr * cr + vi * vg;
                float hn = vo * tanhf_fast(cn);
                h_out[(size_t)Pout * 128 + d] = hn;
                c_out[(size_t)Pout * 128 + d] = cn;
            }
        }
    }
}

// Big streaming output: out[i,0:128] = leaf[i,:], out[i,128:256] = 0 (float4 vectorized).
__global__ __launch_bounds__(256)
void leaves_kernel(const float* __restrict__ leaf, float* __restrict__ out) {
    const size_t total  = (size_t)NRES * 64;     // float4 count for full output
    const size_t stride = (size_t)gridDim.x * blockDim.x;
    for (size_t idx = (size_t)blockIdx.x * blockDim.x + threadIdx.x; idx < total; idx += stride) {
        size_t row = idx >> 6;
        int    c4  = (int)(idx & 63);
        float4 v   = make_float4(0.f, 0.f, 0.f, 0.f);
        if (c4 < 32) v = ((const float4*)leaf)[row * 32 + c4];
        ((float4*)out)[idx] = v;
    }
}

// super_root_vec: mean of 4 tree roots, zero tail.
__global__ void root_kernel(const float* __restrict__ hroots, float* __restrict__ out) {
    int d = threadIdx.x;                         // 128 threads
    float s = 0.25f * (hroots[d] + hroots[128 + d] + hroots[256 + d] + hroots[384 + d]);
    out[d]       = s;
    out[128 + d] = 0.f;
}

extern "C" void kernel_launch(void* const* d_in, const int* in_sizes, int n_in,
                              void* d_out, int out_size, void* d_ws, size_t ws_size,
                              hipStream_t stream) {
    const float* leaf = (const float*)d_in[0];
    const float* W_up = (const float*)d_in[1];
    const float* b_up = (const float*)d_in[2];
    // d_in[3]=W_down, d_in[4]=b_down are dead in the reference; d_in[5]=n_res (constant).
    float* out = (float*)d_out;

    char* ws = (char*)d_ws;
    _Float16* wfrag = (_Float16*)(ws + WF_OFF);
    float* hA = (float*)(ws + HA_OFF);
    float* cA = (float*)(ws + CA_OFF);
    float* hB = (float*)(ws + HB_OFF);
    float* cB = (float*)(ws + CB_OFF);

    // independent streaming output + weight repack
    pack_w_kernel<<<320, 32, 0, stream>>>(W_up, wfrag);
    leaves_kernel<<<65536, 256, 0, stream>>>(leaf, out + 256);

    // 16 up-levels, ping-pong h/c buffers; odd levels write A, even levels write B
    const int parents_per_block = 8 * MTILES_PER_WAVE * 16;   // 512
    const float* hin = leaf;
    const float* cin = nullptr;                  // leaf-level c == 0
    for (int l = 1; l <= 16; ++l) {
        int Mtot = 4 << (16 - l);                // parents across all 4 trees
        float* hout = (l & 1) ? hA : hB;
        float* cout_ = (l & 1) ? cA : cB;
        dim3 grid((Mtot + parents_per_block - 1) / parents_per_block, 8);
        tree_lstm_level_kernel<<<grid, 256, 0, stream>>>(hin, cin, wfrag, b_up, hout, cout_, Mtot);
        hin = hout; cin = cout_;
    }

    // level 16 output lives in the B buffer: 4 roots x 128
    root_kernel<<<1, 128, 0, stream>>>(hin, out);
}
